// NTC_30202210025785
// MI455X (gfx1250) — matrix operand order repaired and checked by reading, not compiled, not run on hardware
//
#include <hip/hip_runtime.h>

// ---------------- problem constants ----------------
#define RANK 16
#define NF 32
#define BATCH 16384
#define WAVES_PER_BLOCK 4
#define THREADS (WAVES_PER_BLOCK * 32)

typedef __attribute__((ext_vector_type(16))) _Float16 v16h;
typedef __attribute__((ext_vector_type(8)))  _Float16 v8h;
typedef __attribute__((ext_vector_type(8)))  float    v8f;
typedef __attribute__((ext_vector_type(4)))  unsigned int u32x4;
typedef __attribute__((ext_vector_type(8)))  int      i32x8;
typedef __attribute__((ext_vector_type(4)))  int      i32x4;

// ---------------- LDS layout (bytes) ----------------
// K=256 rows padded to 264 halfs (528B = 33*16B: keeps 16B alignment for
// v8h loads and decorrelates LDS banks).
#define WROW 264
#define OFF_W1 0
#define OFF_W2 (OFF_W1 + 32 * WROW * 2)   // 16896
#define OFF_W3 (OFF_W2 + 32 * WROW * 2)   // 33792
#define OFF_W0 (OFF_W3 + 32 * WROW * 2)   // 50688  (32x8 f32)
#define OFF_B0 (OFF_W0 + 256 * 4)         // 51712  biases, 32 f32 each
#define OFF_B1 (OFF_B0 + 128)
#define OFF_B2 (OFF_B1 + 128)
#define OFF_B3 (OFF_B2 + 128)
#define OFF_FC (OFF_B3 + 128)             // 33 f32 (fc_w + fc_b)
#define SMEM_SHARED (OFF_FC + 160)        // 52384

// per-wave staging
#define PW_ABC 0                           // a,b,c : 48 f32
#define PW_B1  192                         // conv1 B: 64 cols x 264 halfs
#define PW_B2  (PW_B1 + 64 * WROW * 2)     // +33792
#define PW_B3  (PW_B2 + 16 * WROW * 2)     // +8448 ; conv3 B: 256 f32
#define PW_STRIDE (PW_B3 + 1024)           // 43456
#define SMEM_TOTAL (SMEM_SHARED + WAVES_PER_BLOCK * PW_STRIDE)  // 226208 B

static_assert(SMEM_TOTAL <= 320 * 1024, "exceeds CDNA5 WGP LDS");
static_assert(PW_B2 - PW_B1 >= 32 * 256 * 4 / 4, "TDM staging fits in B1");

#if defined(__gfx1250__) && __has_builtin(__builtin_amdgcn_tensor_load_to_lds)
#define USE_TDM 1
#else
#define USE_TDM 0
#endif

__device__ __forceinline__ v16h cat8(v8h lo, v8h hi) {
  return __builtin_shufflevector(lo, hi, 0, 1, 2, 3, 4, 5, 6, 7,
                                 8, 9, 10, 11, 12, 13, 14, 15);
}
__device__ __forceinline__ v8f wmma16(v16h a, v16h b, v8f c) {
  // D = A(16x32 f16) * B(32x16 f16) + C(16x16 f32)
  return __builtin_amdgcn_wmma_f32_16x16x32_f16(
      false, a, false, b, (short)0, c, false, false);
}
__device__ __forceinline__ float relu(float x) { return x > 0.f ? x : 0.f; }

#if USE_TDM
// 1-row TDM copy: nelem f32 from global -> LDS byte offset (D# per ISA 8.3/8.4)
// clang-23 toolchain signature: (u32x4 g0, i32x8 g1, i32x4 g2, i32x4 g3,
//                                i32x8 spare, i32 cpol)
__device__ __forceinline__ void tdm_load_f32(const float* src,
                                             unsigned lds_bytes, int nelem) {
  const unsigned long long ga = (unsigned long long)(uintptr_t)src;
  u32x4 g0;
  g0[0] = 1u;                                   // count=1 (valid user D#)
  g0[1] = lds_bytes;                            // D#.lds_addr
  g0[2] = (unsigned)(ga & 0xFFFFFFFFu);         // global_addr[31:0]
  g0[3] = (unsigned)((ga >> 32) & 0x01FFFFFFu)  // global_addr[56:32]
          | 0x80000000u;                        // type=2 ("image")
  i32x8 g1;
  g1[0] = (int)(2u << 16);                      // data_size=4B, no multicast
  g1[1] = (int)(((unsigned)nelem & 0xFFFFu) << 16);  // tensor_dim0 lo16
  g1[2] = (int)(((unsigned)nelem >> 16) | (1u << 16)); // dim0 hi | dim1=1
  g1[3] = (int)(((unsigned)nelem & 0xFFFFu) << 16);  // tile_dim0 = nelem
  g1[4] = 0;                                    // tile_dim1/2 unused
  g1[5] = nelem;                                // tensor_dim0_stride
  g1[6] = 0;
  g1[7] = 0;
  const i32x4 z4 = {0, 0, 0, 0};
  const i32x8 z8 = {0, 0, 0, 0, 0, 0, 0, 0};
  __builtin_amdgcn_tensor_load_to_lds(g0, g1, z4, z4, z8, 0);
}
#endif

__global__ void __launch_bounds__(THREADS)
ntc_fused_kernel(const int* __restrict__ gi, const int* __restrict__ gj,
                 const int* __restrict__ gk, const float* __restrict__ gA,
                 const float* __restrict__ gB, const float* __restrict__ gC,
                 const float* __restrict__ gW0, const float* __restrict__ gb0,
                 const float* __restrict__ gW1, const float* __restrict__ gb1,
                 const float* __restrict__ gW2, const float* __restrict__ gb2,
                 const float* __restrict__ gW3, const float* __restrict__ gb3,
                 const float* __restrict__ gfcw, const float* __restrict__ gfcb,
                 float* __restrict__ out) {
  extern __shared__ char smem[];
  const int tid = threadIdx.x;
  const int lane = tid & 31;
  const int wave = tid >> 5;

  _Float16* sW1 = (_Float16*)(smem + OFF_W1);
  _Float16* sW2 = (_Float16*)(smem + OFF_W2);
  _Float16* sW3 = (_Float16*)(smem + OFF_W3);
  float* sW0 = (float*)(smem + OFF_W0);
  float* sb0 = (float*)(smem + OFF_B0);
  float* sb1 = (float*)(smem + OFF_B1);
  float* sb2 = (float*)(smem + OFF_B2);
  float* sb3 = (float*)(smem + OFF_B3);
  float* sfc = (float*)(smem + OFF_FC);  // [0..31] fc_w, [32] fc_b

  // ---- early: load gather indices, prefetch the HBM rows so the gather
  //      latency overlaps weight staging ----
  const int sample = blockIdx.x * WAVES_PER_BLOCK + wave;
  const int ii = gi[sample], jj = gj[sample], kk = gk[sample];
  __builtin_prefetch(gA + ii * RANK, 0, 3);
  __builtin_prefetch(gB + jj * RANK, 0, 3);
  __builtin_prefetch(gC + kk * RANK, 0, 3);

  // ---- stage 0: weight staging (once per block) ----
#if USE_TDM
  // DMA W1/W2/W3 (f32) into the (still unused) per-wave B1 regions via the
  // Tensor Data Mover, then convert to padded-f16 GEMM layout.
  if (wave == 0) {
    tdm_load_f32(gW1, SMEM_SHARED + 0 * PW_STRIDE + PW_B1, 32 * 256);
    tdm_load_f32(gW2, SMEM_SHARED + 1 * PW_STRIDE + PW_B1, 32 * 256);
    tdm_load_f32(gW3, SMEM_SHARED + 2 * PW_STRIDE + PW_B1, 32 * 256);
    __builtin_amdgcn_s_wait_tensorcnt(0);
  }
#endif
  if (tid < 256) sW0[tid] = gW0[tid];
  if (tid < 32) {
    sb0[tid] = gb0[tid];
    sb1[tid] = gb1[tid];
    sb2[tid] = gb2[tid];
    sb3[tid] = gb3[tid];
    sfc[tid] = gfcw[tid];
  }
  if (tid == 0) sfc[32] = gfcb[0];
#if USE_TDM
  __syncthreads();  // TDM data visible to all waves
  {
    const float* t1 = (const float*)(smem + SMEM_SHARED + 0 * PW_STRIDE + PW_B1);
    const float* t2 = (const float*)(smem + SMEM_SHARED + 1 * PW_STRIDE + PW_B1);
    const float* t3 = (const float*)(smem + SMEM_SHARED + 2 * PW_STRIDE + PW_B1);
    for (int idx = tid; idx < 32 * 256; idx += THREADS) {
      const int r = idx >> 8, k = idx & 255;
      sW1[r * WROW + k] = (_Float16)t1[idx];
      sW2[r * WROW + k] = (_Float16)t2[idx];
      sW3[r * WROW + k] = (_Float16)t3[idx];
    }
  }
#else
  for (int idx = tid; idx < 32 * 256; idx += THREADS) {
    const int r = idx >> 8, k = idx & 255;
    sW1[r * WROW + k] = (_Float16)gW1[idx];
    sW2[r * WROW + k] = (_Float16)gW2[idx];
    sW3[r * WROW + k] = (_Float16)gW3[idx];
  }
#endif
  __syncthreads();

  // ---- per-wave sample ----
  char* wbase = smem + SMEM_SHARED + wave * PW_STRIDE;
  float* abc = (float*)(wbase + PW_ABC);
  _Float16* sB1 = (_Float16*)(wbase + PW_B1);
  _Float16* sB2 = (_Float16*)(wbase + PW_B2);
  float* sB3 = (float*)(wbase + PW_B3);

  // gather a = A[i], b = B[j], c = C[k]  (lanes 0..15; rows are L2-warm)
  if (lane < RANK) {
    abc[lane] = gA[ii * RANK + lane];
    abc[16 + lane] = gB[jj * RANK + lane];
    abc[32 + lane] = gC[kk * RANK + lane];
  }
  // DS ops from one wave execute in order; loads below see the stores.

  // ---- stage 1: conv0 via rank-1 factorization (VALU) ----
  // x0[f,R,S,T] = relu(b0[f] + a[2R]*h0[S,T] + a[2R+1]*h1[S,T]),
  // hd[S,T]     = sum_{ds',dt'} W0[f,d,ds',dt'] * b[2S+ds'] * c[2T+dt'].
  // lane == filter f. Iterate over conv1 *patches* (R1,S1,T1) so the 8 taps
  // of a patch (k = f*8 + tap) are contiguous -> one ds_store_b128 each.
  {
    float ar[16];
#pragma unroll
    for (int r = 0; r < 16; ++r) ar[r] = abc[r];
    float w0c[8];
#pragma unroll
    for (int t = 0; t < 8; ++t) w0c[t] = sW0[lane * 8 + t];
    const float bias0 = sb0[lane];

    for (int S1 = 0; S1 < 4; ++S1) {
      float bq[4];
#pragma unroll
      for (int u = 0; u < 4; ++u) bq[u] = abc[16 + 4 * S1 + u];
      for (int T1 = 0; T1 < 4; ++T1) {
        float cq[4];
#pragma unroll
        for (int u = 0; u < 4; ++u) cq[u] = abc[32 + 4 * T1 + u];
        float h[2][2][2];  // h[d][ds][dt] for S=2*S1+ds, T=2*T1+dt
#pragma unroll
        for (int ds = 0; ds < 2; ++ds)
#pragma unroll
          for (int dt = 0; dt < 2; ++dt) {
            const float p00 = bq[2 * ds] * cq[2 * dt];
            const float p01 = bq[2 * ds] * cq[2 * dt + 1];
            const float p10 = bq[2 * ds + 1] * cq[2 * dt];
            const float p11 = bq[2 * ds + 1] * cq[2 * dt + 1];
            h[0][ds][dt] = w0c[0] * p00 + w0c[1] * p01 + w0c[2] * p10 + w0c[3] * p11;
            h[1][ds][dt] = w0c[4] * p00 + w0c[5] * p01 + w0c[6] * p10 + w0c[7] * p11;
          }
#pragma unroll
        for (int R1 = 0; R1 < 4; ++R1) {
          v8h pk;
#pragma unroll
          for (int dr = 0; dr < 2; ++dr) {
            const float a0 = ar[4 * R1 + 2 * dr];
            const float a1 = ar[4 * R1 + 2 * dr + 1];
#pragma unroll
            for (int ds = 0; ds < 2; ++ds)
#pragma unroll
              for (int dt = 0; dt < 2; ++dt) {
                const float v =
                    relu(fmaf(a0, h[0][ds][dt], fmaf(a1, h[1][ds][dt], bias0)));
                pk[dr * 4 + ds * 2 + dt] = (_Float16)v;
              }
          }
          const int col = R1 * 16 + S1 * 4 + T1;
          *(v8h*)(&sB1[col * WROW + lane * 8]) = pk;  // 16B-aligned b128 store
        }
      }
    }
  }

  const int nc = lane & 15;   // column within 16-wide N tile / A row
  const int hi = lane >> 4;   // lane half (selects K sub-chunks)

  // ---- stage 2: conv1 as WMMA GEMM  W1(32x256) x B1(256x64) ----
  // ks-outer: A fragments loaded once per K-step and reused by all 4 N-tiles
  // (64 accumulator VGPRs live; occupancy is LDS-bound anyway).
  {
    const v8f vzero = {0.f, 0.f, 0.f, 0.f, 0.f, 0.f, 0.f, 0.f};
    v8f acc[4][2];
#pragma unroll
    for (int nt = 0; nt < 4; ++nt) {
      acc[nt][0] = vzero;
      acc[nt][1] = vzero;
    }
#pragma unroll
    for (int ks = 0; ks < 8; ++ks) {
      // A: row = nc (+16 tile1); lane half holds K = ks*32+hi*8+{0..7,16..23}
      const int kA = ks * 32 + (hi << 3);
      const _Float16* ap0 = &sW1[nc * WROW + kA];
      const v16h af0 = cat8(*(const v8h*)ap0, *(const v8h*)(ap0 + 16));
      const _Float16* ap1 = &sW1[(nc + 16) * WROW + kA];
      const v16h af1 = cat8(*(const v8h*)ap1, *(const v8h*)(ap1 + 16));
#pragma unroll
      for (int nt = 0; nt < 4; ++nt) {
        // B: lane half holds K = ks*32+hi*8+{0..7,16..23}, col = nt*16+nc
        const _Float16* bp = &sB1[(nt * 16 + nc) * WROW + ks * 32 + (hi << 3)];
        const v16h bf = cat8(*(const v8h*)bp, *(const v8h*)(bp + 16));
        acc[nt][0] = wmma16(af0, bf, acc[nt][0]);
        acc[nt][1] = wmma16(af1, bf, acc[nt][1]);
      }
    }
    // bias+ReLU, scatter into conv2's im2col B (stride==kernel: permutation)
#pragma unroll
    for (int nt = 0; nt < 4; ++nt) {
      const int p = nt * 16 + nc;  // x1 position R1*16+S1*4+T1
      const int R1 = p >> 4, S1 = (p >> 2) & 3, T1 = p & 3;
      const int q = ((R1 >> 1) << 2) + ((S1 >> 1) << 1) + (T1 >> 1);
      const int tap = ((R1 & 1) << 2) + ((S1 & 1) << 1) + (T1 & 1);
#pragma unroll
      for (int g = 0; g < 8; ++g) {
        const int f0 = g + (hi << 3);  // C/D layout: M = g + 8*hi
        const int f1 = f0 + 16;
        sB2[q * WROW + f0 * 8 + tap] = (_Float16)relu(acc[nt][0][g] + sb1[f0]);
        sB2[q * WROW + f1 * 8 + tap] = (_Float16)relu(acc[nt][1][g] + sb1[f1]);
      }
    }
  }

  // ---- stage 3: conv2 as WMMA GEMM  W2(32x256) x B2(256x8, N padded 16) ----
  {
    v8f acc0 = {0.f, 0.f, 0.f, 0.f, 0.f, 0.f, 0.f, 0.f};
    v8f acc1 = acc0;
#pragma unroll
    for (int ks = 0; ks < 8; ++ks) {
      const _Float16* bp = &sB2[nc * WROW + ks * 32 + (hi << 3)];
      const v16h bf = cat8(*(const v8h*)bp, *(const v8h*)(bp + 16));
      const int kA = ks * 32 + (hi << 3);
      const _Float16* ap0 = &sW2[nc * WROW + kA];
      const v16h af0 = cat8(*(const v8h*)ap0, *(const v8h*)(ap0 + 16));
      const _Float16* ap1 = &sW2[(nc + 16) * WROW + kA];
      const v16h af1 = cat8(*(const v8h*)ap1, *(const v8h*)(ap1 + 16));
      acc0 = wmma16(af0, bf, acc0);
      acc1 = wmma16(af1, bf, acc1);
    }
    const int q2 = nc;          // conv2 output position (valid < 8)
    const bool valid = q2 < 8;  // columns 8..15 are padding
#pragma unroll
    for (int g = 0; g < 8; ++g) {
      const int f0 = g + (hi << 3);
      const int f1 = f0 + 16;
      const float v0 = relu(acc0[g] + sb2[f0]);
      const float v1 = relu(acc1[g] + sb2[f1]);
      if (valid) {
        sB3[f0 * 8 + q2] = v0;  // conv3 K index = f*8 + q (q == tap)
        sB3[f1 * 8 + q2] = v1;
      }
    }
  }

  // ---- stage 4: conv3 (32x256 x 256x1) on VALU, lane == out filter ----
  float acc = sb3[lane];
  {
    const _Float16* wr = &sW3[lane * WROW];
#pragma unroll 8
    for (int k = 0; k < 256; ++k) acc = fmaf((float)wr[k], sB3[k], acc);
  }
  const float y = relu(acc);

  // ---- stage 5: FC + sigmoid, wave32 butterfly reduction ----
  float part = y * sfc[lane];
#pragma unroll
  for (int off = 16; off > 0; off >>= 1) part += __shfl_xor(part, off, 32);
  if (lane == 0) {
    const float z = part + sfc[32];
    out[sample] = 1.f / (1.f + __expf(-z));
  }
}

extern "C" void kernel_launch(void* const* d_in, const int* in_sizes, int n_in,
                              void* d_out, int out_size, void* d_ws,
                              size_t ws_size, hipStream_t stream) {
  (void)in_sizes; (void)n_in; (void)d_ws; (void)ws_size; (void)out_size;
  const int* gi = (const int*)d_in[0];
  const int* gj = (const int*)d_in[1];
  const int* gk = (const int*)d_in[2];
  const float* gA = (const float*)d_in[3];
  const float* gB = (const float*)d_in[4];
  const float* gC = (const float*)d_in[5];
  const float* gW0 = (const float*)d_in[6];
  const float* gb0 = (const float*)d_in[7];
  const float* gW1 = (const float*)d_in[8];
  const float* gb1 = (const float*)d_in[9];
  const float* gW2 = (const float*)d_in[10];
  const float* gb2 = (const float*)d_in[11];
  const float* gW3 = (const float*)d_in[12];
  const float* gb3 = (const float*)d_in[13];
  const float* gfcw = (const float*)d_in[14];
  const float* gfcb = (const float*)d_in[15];
  float* out = (float*)d_out;

  (void)hipFuncSetAttribute((const void*)ntc_fused_kernel,
                            hipFuncAttributeMaxDynamicSharedMemorySize,
                            SMEM_TOTAL);
  ntc_fused_kernel<<<BATCH / WAVES_PER_BLOCK, THREADS, SMEM_TOTAL, stream>>>(
      gi, gj, gk, gA, gB, gC, gW0, gb0, gW1, gb1, gW2, gb2, gW3, gb3, gfcw,
      gfcb, out);
}
